// LinearAttention_2001454760421
// MI455X (gfx1250) — compile-verified, matrix operand order checked by reading, not run on hardware
//
#include <hip/hip_runtime.h>
#include <hip/hip_bf16.h>
#include <math.h>

// ---------------- problem constants ----------------
#define BB    4
#define NTOK  4096
#define DIMD  512
#define HH    8
#define CC    64
#define HID   512          // HH*CC
#define NQKV  1536         // 3*HID
#define NMEMT 8
#define MROWS (BB*NTOK)    // 16384
#define SQRT_DIM 22.627416997969522f   // sqrt(512)

typedef __bf16 bf16;
typedef __bf16 v16bf __attribute__((ext_vector_type(16)));
typedef __bf16 v8bf  __attribute__((ext_vector_type(8)));
typedef float  v8f   __attribute__((ext_vector_type(8)));

static __device__ __forceinline__ v8f v8f_zero() {
    v8f z = {0.f,0.f,0.f,0.f,0.f,0.f,0.f,0.f};
    return z;
}

#define WMMA_BF16(A_, B_, C_) \
    __builtin_amdgcn_wmma_f32_16x16x32_bf16(false, (A_), false, (B_), (short)0, (C_), false, false)

// A-fragment loader per CDNA5 ISA 16-bit A layout (16x32, MxK):
//   lanes 0-15 : M = lane,    K = {k+0..7,  k+16..23}
//   lanes 16-31: M = lane-16, K = {k+8..15, k+24..31}
// Works for B fragments too when B is stored K-contiguous per output column.
static __device__ __forceinline__ v16bf load_frag_rowmajor(const bf16* base, int ldk, int lane) {
    int m      = lane & 15;
    int khalf  = (lane >> 4) * 8;              // 0 or 8
    const bf16* p = base + (size_t)m * ldk + khalf;
    v8bf lo = *(const v8bf*)(p);               // K khalf..khalf+7
    v8bf hi = *(const v8bf*)(p + 16);          // K khalf+16..khalf+23
    return __builtin_shufflevector(lo, hi, 0,1,2,3,4,5,6,7,8,9,10,11,12,13,14,15);
}

// C/D layout: vgpr i -> row = 8*(lane>=16)+i, col = lane&15
static __device__ __forceinline__ void store_c_bf16(bf16* base, int ldn, int lane, v8f c) {
    int n  = lane & 15;
    int mh = (lane >> 4) * 8;
#pragma unroll
    for (int i = 0; i < 8; ++i)
        base[(size_t)(mh + i) * ldn + n] = (bf16)c[i];
}

// Software-pipelined 32x64 wave GEMM: A panel (32 rows) in LDS, B panel in global
// (BT, K-contiguous). Each K-step: 2 ds A-frag loads + 4 global B-frag loads feed
// 8 WMMAs (2x the matrix-op/load ratio of a 16x64 tile), and the next step's B
// fragments are requested before the current WMMAs so loadcnt waits can be partial.
template <int KTOT>
static __device__ __forceinline__ void wave_gemm_32x64(const bf16* __restrict__ Ash,
                                                       const bf16* __restrict__ BTp,
                                                       int lane, v8f acc[2][4]) {
    v16bf a0 = load_frag_rowmajor(Ash, KTOT, lane);
    v16bf a1 = load_frag_rowmajor(Ash + 16 * KTOT, KTOT, lane);
    v16bf b[4];
#pragma unroll
    for (int j = 0; j < 4; ++j)
        b[j] = load_frag_rowmajor(BTp + (size_t)j * 16 * KTOT, KTOT, lane);
#pragma unroll
    for (int k0 = 0; k0 < KTOT; k0 += 32) {
        v16bf a0n, a1n;
        v16bf bn[4];
        if (k0 + 32 < KTOT) {
            a0n = load_frag_rowmajor(Ash + k0 + 32, KTOT, lane);
            a1n = load_frag_rowmajor(Ash + 16 * KTOT + k0 + 32, KTOT, lane);
            __builtin_prefetch(BTp + k0 + 96, 0, 1);
#pragma unroll
            for (int j = 0; j < 4; ++j)
                bn[j] = load_frag_rowmajor(BTp + (size_t)j * 16 * KTOT + k0 + 32, KTOT, lane);
        }
#pragma unroll
        for (int j = 0; j < 4; ++j) acc[0][j] = WMMA_BF16(a0, b[j], acc[0][j]);
#pragma unroll
        for (int j = 0; j < 4; ++j) acc[1][j] = WMMA_BF16(a1, b[j], acc[1][j]);
        if (k0 + 32 < KTOT) {
            a0 = a0n; a1 = a1n;
#pragma unroll
            for (int j = 0; j < 4; ++j) b[j] = bn[j];
        }
    }
}

// ---------------- K0: f32 weight -> bf16 transpose ----------------
__global__ void k_transpose_bf16(const float* __restrict__ src, bf16* __restrict__ dst,
                                 int rows, int cols) {
    int idx = blockIdx.x * blockDim.x + threadIdx.x;
    if (idx >= rows * cols) return;
    int r = idx / cols, c = idx % cols;
    dst[(size_t)c * rows + r] = (bf16)src[idx];
}

// ---------------- K1: RMSNorm(x) -> bf16 ----------------
__global__ void k_rmsnorm_bf16(const float* __restrict__ x, const float* __restrict__ w,
                               bf16* __restrict__ out) {
    __shared__ float sred[8];
    int row = blockIdx.x;
    const float* xr = x + (size_t)row * DIMD;
    float ss = 0.f;
    for (int c = threadIdx.x; c < DIMD; c += blockDim.x) { float v = xr[c]; ss += v * v; }
    for (int off = 16; off; off >>= 1) ss += __shfl_down(ss, off, 32);
    if ((threadIdx.x & 31) == 0) sred[threadIdx.x >> 5] = ss;
    __syncthreads();
    if (threadIdx.x == 0) { float t = 0.f; for (int i = 0; i < 8; ++i) t += sred[i]; sred[0] = t; }
    __syncthreads();
    float s = SQRT_DIM / fmaxf(sqrtf(sred[0]), 1e-12f);
    bf16* orow = out + (size_t)row * DIMD;
    for (int c = threadIdx.x; c < DIMD; c += blockDim.x)
        orow[c] = (bf16)(xr[c] * s * w[c]);
}

// ---------------- K2: QKV GEMM (bf16 WMMA, 32-row panel, pipelined) ----------------
__global__ void k_gemm_qkv(const bf16* __restrict__ A, const bf16* __restrict__ BT,
                           bf16* __restrict__ Cout) {
    __shared__ __align__(16) bf16 Ash[32 * DIMD];      // 32 KB
    int m0 = blockIdx.x * 32;
    {   // cooperative 32x512 A-panel load (b128 global -> b128 ds)
        const v8bf* src = (const v8bf*)(A + (size_t)m0 * DIMD);
        v8bf* dst = (v8bf*)Ash;
        for (int i = threadIdx.x; i < 32 * DIMD / 8; i += blockDim.x) dst[i] = src[i];
    }
    __syncthreads();
    int wave = threadIdx.x >> 5, lane = threadIdx.x & 31;
    int n0 = blockIdx.y * 512 + wave * 64;
    v8f acc[2][4] = {{v8f_zero(), v8f_zero(), v8f_zero(), v8f_zero()},
                     {v8f_zero(), v8f_zero(), v8f_zero(), v8f_zero()}};
    wave_gemm_32x64<DIMD>(Ash, BT + (size_t)n0 * DIMD, lane, acc);
#pragma unroll
    for (int s = 0; s < 2; ++s)
#pragma unroll
        for (int j = 0; j < 4; ++j)
            store_c_bf16(Cout + (size_t)(m0 + s * 16) * NQKV + (n0 + 16 * j), NQKV, lane, acc[s][j]);
}

// ---------------- K3a: k-softmax stats over tokens (incl. mem tokens) ----------------
__global__ void k_kstat(const bf16* __restrict__ qkv, const float* __restrict__ memk,
                        float* __restrict__ kmax, float* __restrict__ ksum) {
    __shared__ float sred[8];
    int row = blockIdx.x;                       // B*H*C = 2048 rows
    int b = row / (HH * CC), hc = row % (HH * CC), h = hc / CC, c = hc % CC;
    const bf16* kp = qkv + (size_t)b * NTOK * NQKV + HID + h * CC + c;
    float mx = -1e30f;
    for (int t = threadIdx.x; t < NTOK; t += blockDim.x)
        mx = fmaxf(mx, (float)kp[(size_t)t * NQKV]);
    if (threadIdx.x < NMEMT) mx = fmaxf(mx, memk[(h * CC + c) * NMEMT + threadIdx.x]);
    for (int off = 16; off; off >>= 1) mx = fmaxf(mx, __shfl_down(mx, off, 32));
    if ((threadIdx.x & 31) == 0) sred[threadIdx.x >> 5] = mx;
    __syncthreads();
    if (threadIdx.x == 0) { float m = sred[0]; for (int i = 1; i < 8; ++i) m = fmaxf(m, sred[i]); sred[0] = m; }
    __syncthreads();
    float gmax = sred[0];
    __syncthreads();
    float sum = 0.f;
    for (int t = threadIdx.x; t < NTOK; t += blockDim.x)
        sum += expf((float)kp[(size_t)t * NQKV] - gmax);
    if (threadIdx.x < NMEMT) sum += expf(memk[(h * CC + c) * NMEMT + threadIdx.x] - gmax);
    for (int off = 16; off; off >>= 1) sum += __shfl_down(sum, off, 32);
    if ((threadIdx.x & 31) == 0) sred[threadIdx.x >> 5] = sum;
    __syncthreads();
    if (threadIdx.x == 0) {
        float t = 0.f; for (int i = 0; i < 8; ++i) t += sred[i];
        kmax[row] = gmax; ksum[row] = t;
    }
}

// ---------------- K3b: context = softmax(k) @ v^T per (b,h) ----------------
// Double-buffered LDS tiles; result written TRANSPOSED as bf16 (ctxT[e][d]) so the
// consumer (K4) gets vectorized b128 B-fragment loads.
#define LDT 40   // LDS tile row stride (mult of 8 -> 16B aligned fragment groups)
__global__ void k_context(const bf16* __restrict__ qkv, const float* __restrict__ memkv,
                          const float* __restrict__ kmax, const float* __restrict__ ksum,
                          bf16* __restrict__ ctxT) {
    __shared__ __align__(16) bf16 Ksh[2][CC * LDT];   // [buf][d][token]
    __shared__ __align__(16) bf16 Vsh[2][CC * LDT];   // [buf][e][token]
    int bh = blockIdx.x;                 // 32 blocks
    int bb = bh / HH, h = bh % HH;
    int wave = threadIdx.x >> 5, lane = threadIdx.x & 31;
    int d0 = (wave & 3) * 16;
    int e0 = (wave >> 2) * 32;
    // loader mapping: 32 tokens x 8 d-groups per 256-thread block
    int tt = threadIdx.x >> 3;
    int dd = (threadIdx.x & 7) * 8;
    float km[8], rs[8];
#pragma unroll
    for (int i = 0; i < 8; ++i) {
        int r = (bb * HH + h) * CC + dd + i;
        km[i] = kmax[r];
        rs[i] = 1.f / ksum[r];
    }
    // preload tile 0
    {
        size_t rowoff = (size_t)(bb * NTOK + tt) * NQKV + h * CC + dd;
        v8bf kv8 = *(const v8bf*)(qkv + rowoff + HID);
        v8bf vv8 = *(const v8bf*)(qkv + rowoff + 2 * HID);
#pragma unroll
        for (int i = 0; i < 8; ++i) {
            Ksh[0][(dd + i) * LDT + tt] = (bf16)(expf((float)kv8[i] - km[i]) * rs[i]);
            Vsh[0][(dd + i) * LDT + tt] = vv8[i];
        }
    }
    v8f acc0 = v8f_zero(), acc1 = v8f_zero();
    int cur = 0;
    for (int n0 = 0; n0 < NTOK; n0 += 32) {
        __syncthreads();                       // buf[cur] ready for all waves
        bool have = (n0 + 32) < NTOK;
        v8bf kn8, vn8;
        if (have) {                            // issue next tile's global loads early
            size_t rowoff = (size_t)(bb * NTOK + n0 + 32 + tt) * NQKV + h * CC + dd;
            kn8 = *(const v8bf*)(qkv + rowoff + HID);
            vn8 = *(const v8bf*)(qkv + rowoff + 2 * HID);
        }
        v16bf a  = load_frag_rowmajor(Ksh[cur] + d0 * LDT, LDT, lane);
        v16bf b0 = load_frag_rowmajor(Vsh[cur] + e0 * LDT, LDT, lane);
        v16bf b1 = load_frag_rowmajor(Vsh[cur] + (e0 + 16) * LDT, LDT, lane);
        acc0 = WMMA_BF16(a, b0, acc0);
        acc1 = WMMA_BF16(a, b1, acc1);
        if (have) {                            // fill the other buffer behind the WMMAs
            int nxt = cur ^ 1;
#pragma unroll
            for (int i = 0; i < 8; ++i) {
                Ksh[nxt][(dd + i) * LDT + tt] = (bf16)(expf((float)kn8[i] - km[i]) * rs[i]);
                Vsh[nxt][(dd + i) * LDT + tt] = vn8[i];
            }
        }
        cur ^= 1;
    }
    // rank-8 mem-token correction in f32, then store context transposed as bf16
    const float* memk = memkv;
    const float* memv = memkv + HH * CC * NMEMT;
    int nb = lane & 15, mh = (lane >> 4) * 8;
    v8bf s0, s1;
#pragma unroll
    for (int i = 0; i < 8; ++i) {
        int d = d0 + mh + i;
        int r = (bb * HH + h) * CC + d;
        float kmv = kmax[r], rsv = 1.f / ksum[r];
        float add0 = 0.f, add1 = 0.f;
#pragma unroll
        for (int t = 0; t < NMEMT; ++t) {
            float aw = expf(memk[(h * CC + d) * NMEMT + t] - kmv) * rsv;
            add0 += aw * memv[(h * CC + e0 + nb) * NMEMT + t];
            add1 += aw * memv[(h * CC + e0 + 16 + nb) * NMEMT + t];
        }
        s0[i] = (bf16)(acc0[i] + add0);
        s1[i] = (bf16)(acc1[i] + add1);
    }
    size_t rbase = (size_t)(bb * HH + h) * CC * CC;
    *(v8bf*)(ctxT + rbase + (size_t)(e0 + nb) * CC + d0 + mh)      = s0;   // ctxT[e][d]
    *(v8bf*)(ctxT + rbase + (size_t)(e0 + 16 + nb) * CC + d0 + mh) = s1;
}

// ---------------- K4: out = ctx^T @ softmax_d(q)*scale, per head, WMMA ----------------
__global__ void k_attn_out(const bf16* __restrict__ qkv, const bf16* __restrict__ ctxT,
                           bf16* __restrict__ attn) {
    __shared__ __align__(16) bf16 Qsh[HH * 16 * CC];   // per-wave(head) [token][d]
    int lane = threadIdx.x & 31, h = threadIdx.x >> 5;  // wave == head
    int row0 = blockIdx.x * 16;
    int bb = row0 >> 12;
    int t = lane & 15, half = lane >> 4;
    // feature softmax over d=64 for token t (lane pair t / t+16 split d halves)
    const bf16* qp = qkv + (size_t)(row0 + t) * NQKV + h * CC + half * 32;
    float qv[32];
    float mx = -1e30f;
#pragma unroll
    for (int i = 0; i < 32; ++i) { qv[i] = (float)qp[i]; mx = fmaxf(mx, qv[i]); }
    mx = fmaxf(mx, __shfl_xor(mx, 16, 32));
    float sum = 0.f;
#pragma unroll
    for (int i = 0; i < 32; ++i) sum += expf(qv[i] - mx);
    sum += __shfl_xor(sum, 16, 32);
    float sc = 0.125f / sum;                 // scale = C^-0.5 = 0.125
    bf16* qs = Qsh + h * 16 * CC + t * CC + half * 32;
#pragma unroll
    for (int i = 0; i < 32; ++i) qs[i] = (bf16)(expf(qv[i] - mx) * sc);
    __syncthreads();
    const bf16* cb = ctxT + (size_t)(bb * HH + h) * CC * CC;   // [e][d], ldk = 64
    v16bf a0 = load_frag_rowmajor(Qsh + h * 16 * CC, CC, lane);
    v16bf a1 = load_frag_rowmajor(Qsh + h * 16 * CC + 32, CC, lane);
    v8f acc[4] = {v8f_zero(), v8f_zero(), v8f_zero(), v8f_zero()};
#pragma unroll
    for (int j = 0; j < 4; ++j) {
        v16bf b0 = load_frag_rowmajor(cb + (size_t)(j * 16) * CC, CC, lane);
        v16bf b1 = load_frag_rowmajor(cb + (size_t)(j * 16) * CC + 32, CC, lane);
        acc[j] = WMMA_BF16(a0, b0, acc[j]);
        acc[j] = WMMA_BF16(a1, b1, acc[j]);
    }
#pragma unroll
    for (int j = 0; j < 4; ++j)
        store_c_bf16(attn + (size_t)row0 * HID + h * CC + j * 16, HID, lane, acc[j]);
}

// ---------------- K5: final GEMM + bias + fused RMSNorm -> f32 out ----------------
__global__ void k_gemm_out(const bf16* __restrict__ A, const bf16* __restrict__ BT,
                           const float* __restrict__ bias, const float* __restrict__ onw,
                           float* __restrict__ Y) {
    __shared__ __align__(16) bf16 Ash[32 * HID];       // 32 KB
    __shared__ float rowsq[32];
    int m0 = blockIdx.x * 32;
    {
        const v8bf* src = (const v8bf*)(A + (size_t)m0 * HID);
        v8bf* dst = (v8bf*)Ash;
        for (int i = threadIdx.x; i < 32 * HID / 8; i += blockDim.x) dst[i] = src[i];
    }
    if (threadIdx.x < 32) rowsq[threadIdx.x] = 0.f;
    __syncthreads();
    int wave = threadIdx.x >> 5, lane = threadIdx.x & 31;
    int n0 = wave * 64;                         // block owns full 32x512 rows
    v8f acc[2][4] = {{v8f_zero(), v8f_zero(), v8f_zero(), v8f_zero()},
                     {v8f_zero(), v8f_zero(), v8f_zero(), v8f_zero()}};
    wave_gemm_32x64<HID>(Ash, BT + (size_t)n0 * HID, lane, acc);
    int nb = lane & 15, mh = (lane >> 4) * 8;
#pragma unroll
    for (int s = 0; s < 2; ++s) {
        float pr[8] = {0.f,0.f,0.f,0.f,0.f,0.f,0.f,0.f};
#pragma unroll
        for (int j = 0; j < 4; ++j) {
            float bv = bias[n0 + j * 16 + nb];
#pragma unroll
            for (int i = 0; i < 8; ++i) { float v = acc[s][j][i] + bv; acc[s][j][i] = v; pr[i] += v * v; }
        }
#pragma unroll
        for (int i = 0; i < 8; ++i) atomicAdd(&rowsq[s * 16 + mh + i], pr[i]);   // ds_add_f32
    }
    __syncthreads();
#pragma unroll
    for (int s = 0; s < 2; ++s)
#pragma unroll
        for (int i = 0; i < 8; ++i) {
            float sc = SQRT_DIM / fmaxf(sqrtf(rowsq[s * 16 + mh + i]), 1e-12f);
#pragma unroll
            for (int j = 0; j < 4; ++j) {
                int n = n0 + j * 16 + nb;
                Y[(size_t)(m0 + s * 16 + mh + i) * DIMD + n] = acc[s][j][i] * sc * onw[n];
            }
        }
}

// ---------------- host orchestration ----------------
extern "C" void kernel_launch(void* const* d_in, const int* in_sizes, int n_in,
                              void* d_out, int out_size, void* d_ws, size_t ws_size,
                              hipStream_t stream) {
    const float* x      = (const float*)d_in[0];
    const float* norm_w = (const float*)d_in[1];
    const float* w_qkv  = (const float*)d_in[2];
    const float* mem_kv = (const float*)d_in[3];
    const float* w_out  = (const float*)d_in[4];
    const float* b_out  = (const float*)d_in[5];
    const float* onw    = (const float*)d_in[6];
    float* Y = (float*)d_out;

    char* ws = (char*)d_ws;
    size_t o = 0;
    auto take = [&](size_t bytes) { size_t r = o; o = (o + bytes + 255) & ~(size_t)255; return r; };
    bf16*  xn     = (bf16*)(ws + take((size_t)MROWS * DIMD * 2));
    bf16*  wqkvT  = (bf16*)(ws + take((size_t)NQKV * DIMD * 2));
    bf16*  woutT  = (bf16*)(ws + take((size_t)DIMD * HID * 2));
    bf16*  qkv    = (bf16*)(ws + take((size_t)MROWS * NQKV * 2));
    float* kmax   = (float*)(ws + take((size_t)BB * HH * CC * 4));
    float* ksum   = (float*)(ws + take((size_t)BB * HH * CC * 4));
    bf16*  ctxT   = (bf16*)(ws + take((size_t)BB * HH * CC * CC * 2));
    bf16*  attn   = (bf16*)(ws + take((size_t)MROWS * HID * 2));

    // weights -> bf16, transposed so WMMA B fragments are K-contiguous
    k_transpose_bf16<<<(DIMD * NQKV + 255) / 256, 256, 0, stream>>>(w_qkv, wqkvT, DIMD, NQKV);
    k_transpose_bf16<<<(HID * DIMD + 255) / 256, 256, 0, stream>>>(w_out, woutT, HID, DIMD);
    // RMSNorm -> bf16 activations
    k_rmsnorm_bf16<<<MROWS, 256, 0, stream>>>(x, norm_w, xn);
    // QKV projection (32-row panels)
    k_gemm_qkv<<<dim3(MROWS / 32, 3), 256, 0, stream>>>(xn, wqkvT, qkv);
    // token-softmax statistics for k
    k_kstat<<<BB * HH * CC, 256, 0, stream>>>(qkv, mem_kv, kmax, ksum);
    // per-head context (k softmax applied in-flight), transposed bf16 output
    k_context<<<BB * HH, 256, 0, stream>>>(qkv, mem_kv, kmax, ksum, ctxT);
    // feature-softmax q, apply context
    k_attn_out<<<MROWS / 16, 256, 0, stream>>>(qkv, ctxT, attn);
    // output projection + bias + fused RMSNorm (32-row panels)
    k_gemm_out<<<MROWS / 32, 256, 0, stream>>>(attn, woutT, b_out, onw, Y);
}
